// ProbSparseAttention_9345848836747
// MI455X (gfx1250) — matrix operand order repaired and checked by reading, not compiled, not run on hardware
//
#include <hip/hip_runtime.h>
#include <hip/hip_bf16.h>

// ---------------------------------------------------------------------------
// ProbSparse attention block for MI455X (gfx1250, wave32, WMMA).
// 4x 4096x1024x1024 GEMMs via v_wmma_f32_16x16x32_bf16, double-buffered LDS,
// async global->LDS staging (ASYNCcnt path) with sync fallback.
// ---------------------------------------------------------------------------

typedef __attribute__((ext_vector_type(16))) __bf16 v16bf;
typedef __attribute__((ext_vector_type(8)))  float  v8f;
typedef __attribute__((ext_vector_type(4)))  int    v4i;

#define BATCH   2
#define SEQ     2048
#define NHEAD   16
#define DK      64
#define DM      1024
#define NU      40
#define ROWS    (BATCH * SEQ)      // 4096
#define NEG_INF (-3.4e38f)

#if defined(__has_builtin)
#  if __has_builtin(__builtin_amdgcn_global_load_async_to_lds_b128)
#    define USE_ASYNC_LDS 1
#  endif
#  if __has_builtin(__builtin_amdgcn_s_wait_asynccnt)
#    define HAVE_WAIT_ASYNC_BUILTIN 1
#  endif
#endif
#ifndef USE_ASYNC_LDS
#  define USE_ASYNC_LDS 0
#endif
#ifndef HAVE_WAIT_ASYNC_BUILTIN
#  define HAVE_WAIT_ASYNC_BUILTIN 0
#endif

// v4i pointers in explicit address spaces (global = AS1, LDS = AS3)
typedef __attribute__((address_space(1))) v4i GV4i;
typedef __attribute__((address_space(3))) v4i LV4i;

static __device__ __forceinline__ GV4i* as_global_v4(const void* p) {
    return (GV4i*)(unsigned long long)(__SIZE_TYPE__)p;
}
static __device__ __forceinline__ LV4i* as_lds_v4(const void* p) {
    // generic LDS pointer: low 32 bits are the LDS offset (aperture rules)
    return (LV4i*)(unsigned int)(__SIZE_TYPE__)p;
}

static __device__ __forceinline__ void wait_async_lds() {
#if USE_ASYNC_LDS
#  if HAVE_WAIT_ASYNC_BUILTIN
    __builtin_amdgcn_s_wait_asynccnt(0);
#  else
    asm volatile("s_wait_asynccnt 0" ::: "memory");
#  endif
#endif
}

static __device__ __forceinline__ unsigned short f2bf(float f) {
    unsigned int u = __float_as_uint(f);
    u += 0x7fffu + ((u >> 16) & 1u);       // round-to-nearest-even
    return (unsigned short)(u >> 16);
}

// ---------------------------------------------------------------------------
// fp32 -> bf16 elementwise convert
// ---------------------------------------------------------------------------
__global__ __launch_bounds__(256)
void cvt_bf16_kernel(const float* __restrict__ in, unsigned short* __restrict__ out, int n) {
    int i = blockIdx.x * 256 + threadIdx.x;
    if (i < n) out[i] = f2bf(in[i]);
}

// ---------------------------------------------------------------------------
// fp32 [K][N] -> bf16 transposed [N][K], 32x32 LDS tiles (coalesced both ways)
// ---------------------------------------------------------------------------
__global__ __launch_bounds__(256)
void cvt_transpose_kernel(const float* __restrict__ W, unsigned short* __restrict__ Wt) {
    __shared__ unsigned short tile[32][33];
    const int bn = blockIdx.x * 32;   // N base
    const int bk = blockIdx.y * 32;   // K base
    const int tx = threadIdx.x & 31, ty = threadIdx.x >> 5;   // 32 x 8
    for (int i = 0; i < 4; i++) {
        int r = ty + i * 8;
        tile[r][tx] = f2bf(W[(size_t)(bk + r) * DM + bn + tx]);
    }
    __syncthreads();
    for (int i = 0; i < 4; i++) {
        int r = ty + i * 8;
        Wt[(size_t)(bn + r) * DM + bk + tx] = tile[tx][r];
    }
}

// ---------------------------------------------------------------------------
// LayerNorm: one block (256 thr) per row of 1024. Emits bf16 hidden and
// bf16 layer-normed hidden (GEMM A operands).
// ---------------------------------------------------------------------------
__global__ __launch_bounds__(256)
void ln_kernel(const float* __restrict__ x,
               const float* __restrict__ gamma,
               const float* __restrict__ beta,
               unsigned short* __restrict__ h16,
               unsigned short* __restrict__ qn16) {
    __shared__ float red[256];
    const int row = blockIdx.x;
    const int t   = threadIdx.x;
    const float* xr = x + (size_t)row * DM;

    float v0 = xr[t], v1 = xr[t + 256], v2 = xr[t + 512], v3 = xr[t + 768];

    red[t] = v0 + v1 + v2 + v3;
    __syncthreads();
    for (int off = 128; off > 0; off >>= 1) {
        if (t < off) red[t] += red[t + off];
        __syncthreads();
    }
    const float mu = red[0] * (1.0f / DM);
    __syncthreads();

    float d0 = v0 - mu, d1 = v1 - mu, d2 = v2 - mu, d3 = v3 - mu;
    red[t] = d0 * d0 + d1 * d1 + d2 * d2 + d3 * d3;
    __syncthreads();
    for (int off = 128; off > 0; off >>= 1) {
        if (t < off) red[t] += red[t + off];
        __syncthreads();
    }
    const float rstd = rsqrtf(red[0] * (1.0f / DM) + 1e-6f);

    const float vv[4] = {v0, v1, v2, v3};
    for (int i = 0; i < 4; i++) {
        int c = t + i * 256;
        float xn = (vv[i] - mu) * rstd * gamma[c] + beta[c];
        h16[(size_t)row * DM + c]  = f2bf(vv[i]);
        qn16[(size_t)row * DM + c] = f2bf(xn);
    }
}

// ---------------------------------------------------------------------------
// bf16 GEMM via WMMA. C[M,N] = A[M,K] * Bt[N,K]^T (Bt row-major, pre-transposed).
//   block tile 128x128, 256 threads = 8 waves (2x4), wave tile 64x32,
//   per-wave 4x2 accumulators of 16x16x32 bf16 WMMA.
//   Double-buffered LDS (padded to 80B row stride -> b128 frag loads, no bank
//   clash); async global->LDS staging overlapped with WMMA when available.
// mode 0/1: write to [B,H,L,DK] layout (q/k/v), scaled by `scale`.
// mode 2:   row-major out + bias + residual (final projection).
// ---------------------------------------------------------------------------
__global__ __launch_bounds__(256)
void gemm_bf16_wmma_kernel(const unsigned short* __restrict__ A,
                           const unsigned short* __restrict__ Bt,
                           float* __restrict__ C,
                           const float* __restrict__ bias,
                           const float* __restrict__ residual,
                           int M, int N, int K, float scale, int mode) {
    // [buf][row][kpair], row stride 20 dwords = 80B (16B aligned, bank-spread)
    __shared__ __align__(16) unsigned int ldsA[2][128][20];
    __shared__ __align__(16) unsigned int ldsB[2][128][20];

    const int t    = threadIdx.x;
    const int wave = t >> 5;
    const int lane = t & 31;
    const int hl   = lane >> 4;      // half-wave select
    const int l16  = lane & 15;
    const int wm   = wave >> 2;      // 0..1
    const int wn   = wave & 3;       // 0..3
    const int blockM = blockIdx.y * 128;
    const int blockN = blockIdx.x * 128;
    const int Kh = K >> 1;

    const unsigned int* A32 = (const unsigned int*)A;   // packed bf16 pairs
    const unsigned int* B32 = (const unsigned int*)Bt;

    v8f acc[4][2];
    for (int i = 0; i < 4; i++)
        for (int j = 0; j < 2; j++)
            for (int e = 0; e < 8; e++) acc[i][j][e] = 0.0f;

    // Stage one 128x32 tile of A and of Bt into LDS buffer `buf`.
    // 512 16-byte chunks per matrix, 256 threads -> 2 chunks each.
    auto stage = [&](int k0, int buf) {
        const int kb = k0 >> 1;
        for (int j = 0; j < 2; j++) {
            int c = t + 256 * j;
            int r = c >> 2, q = (c & 3) << 2;
#if USE_ASYNC_LDS
            __builtin_amdgcn_global_load_async_to_lds_b128(
                as_global_v4(&A32[(size_t)(blockM + r) * Kh + kb + q]),
                as_lds_v4(&ldsA[buf][r][q]), 0, 0);
            __builtin_amdgcn_global_load_async_to_lds_b128(
                as_global_v4(&B32[(size_t)(blockN + r) * Kh + kb + q]),
                as_lds_v4(&ldsB[buf][r][q]), 0, 0);
#else
            const unsigned int* ga = &A32[(size_t)(blockM + r) * Kh + kb + q];
            const unsigned int* gb = &B32[(size_t)(blockN + r) * Kh + kb + q];
            for (int e = 0; e < 4; e++) ldsA[buf][r][q + e] = ga[e];
            for (int e = 0; e < 4; e++) ldsB[buf][r][q + e] = gb[e];
#endif
        }
    };

    stage(0, 0);
    const int nk = K >> 5;

    for (int ki = 0; ki < nk; ki++) {
        const int buf = ki & 1;
        wait_async_lds();          // tile ki resident (async path)
        __syncthreads();           // visible to all; prev compute done
        if (ki + 1 < nk) stage((ki + 1) << 5, buf ^ 1);

        // ---- fragments (documented wave32 layouts), 2x b128 per fragment
        v16bf afr[4];
        for (int tm = 0; tm < 4; tm++) {
            int row = wm * 64 + tm * 16 + l16;
            uint4 lo = *(const uint4*)&ldsA[buf][row][hl * 4];      // K  0..7 (+hl*8)
            uint4 hi = *(const uint4*)&ldsA[buf][row][8 + hl * 4];  // K 16..23 (+hl*8)
            __builtin_memcpy(&afr[tm], &lo, 16);
            __builtin_memcpy((char*)&afr[tm] + 16, &hi, 16);
        }
        v16bf bfr[2];
        for (int tn = 0; tn < 2; tn++) {
            int col = wn * 32 + tn * 16 + l16;
            uint4 lo = *(const uint4*)&ldsB[buf][col][hl * 4];
            uint4 hi = *(const uint4*)&ldsB[buf][col][8 + hl * 4];
            __builtin_memcpy(&bfr[tn], &lo, 16);
            __builtin_memcpy((char*)&bfr[tn] + 16, &hi, 16);
        }

        for (int tm = 0; tm < 4; tm++)
            for (int tn = 0; tn < 2; tn++)
                acc[tm][tn] = __builtin_amdgcn_wmma_f32_16x16x32_bf16(
                    false, afr[tm], false, bfr[tn], (short)0, acc[tm][tn], false, false);
    }

    __syncthreads();

    // ---- epilogue (C/D layout: VGPR r -> M = hl*8 + r, N = lane%16)
    for (int tm = 0; tm < 4; tm++)
        for (int tn = 0; tn < 2; tn++)
            for (int r = 0; r < 8; r++) {
                int m = blockM + wm * 64 + tm * 16 + hl * 8 + r;
                int n = blockN + wn * 32 + tn * 16 + l16;
                float val = acc[tm][tn][r] * scale;
                if (mode == 2) {
                    C[(size_t)m * N + n] = val + bias[n] + residual[(size_t)m * N + n];
                } else {
                    int b = m >> 11, l = m & (SEQ - 1);
                    int h = n >> 6,  d = n & (DK - 1);
                    C[(((size_t)(b * NHEAD + h)) * SEQ + l) * DK + d] = val;
                }
            }
}

// ---------------------------------------------------------------------------
// Sampled scores: one wave32 per (b,h,l). 40 sampled K rows; butterfly
// reduction across the wave. M = max(dots) - sum(dots)/L.
// ---------------------------------------------------------------------------
__global__ __launch_bounds__(256)
void sample_score_kernel(const float* __restrict__ q,
                         const float* __restrict__ k,
                         const int* __restrict__ idx,
                         float* __restrict__ Mout) {
    int gw   = blockIdx.x * 8 + (threadIdx.x >> 5);   // 0 .. B*H*L-1
    int lane = threadIdx.x & 31;
    int l    = gw & (SEQ - 1);
    int bh   = gw >> 11;

    const float* qrow = q + (size_t)gw * DK;
    float q0 = qrow[lane], q1 = qrow[lane + 32];

    const float* kbase = k + (size_t)bh * SEQ * DK;
    const int* irow = idx + (size_t)l * NU;

    float mx = NEG_INF, sum = 0.0f;
    for (int s = 0; s < NU; s++) {
        const float* kr = kbase + (size_t)irow[s] * DK;
        float p = q0 * kr[lane] + q1 * kr[lane + 32];
        for (int off = 16; off > 0; off >>= 1) p += __shfl_xor(p, off, 32);
        mx = fmaxf(mx, p);
        sum += p;
    }
    if (lane == 0) Mout[gw] = mx - sum * (1.0f / SEQ);
}

// ---------------------------------------------------------------------------
// Top-40 per (b,h): iterative block argmax over 2048 entries in LDS.
// ---------------------------------------------------------------------------
__global__ __launch_bounds__(256)
void topk_kernel(const float* __restrict__ Mbuf, int* __restrict__ Mtop) {
    __shared__ float vals[SEQ];
    __shared__ float rv[256];
    __shared__ int   ri[256];
    const int bh = blockIdx.x, t = threadIdx.x;

    for (int i = t; i < SEQ; i += 256) vals[i] = Mbuf[(size_t)bh * SEQ + i];
    __syncthreads();

    for (int it = 0; it < NU; it++) {
        float best = NEG_INF; int bi = 0;
        for (int i = t; i < SEQ; i += 256) {
            float v = vals[i];
            if (v > best) { best = v; bi = i; }
        }
        rv[t] = best; ri[t] = bi;
        __syncthreads();
        for (int off = 128; off > 0; off >>= 1) {
            if (t < off && rv[t + off] > rv[t]) { rv[t] = rv[t + off]; ri[t] = ri[t + off]; }
            __syncthreads();
        }
        if (t == 0) {
            Mtop[bh * NU + it] = ri[0];
            vals[ri[0]] = NEG_INF;
        }
        __syncthreads();
    }
}

// ---------------------------------------------------------------------------
// Dense attention for selected queries: one block per (b,h,u).
// scores over all 2048 keys -> softmax -> attn @ v  (k/v stay L2-resident).
// ---------------------------------------------------------------------------
__global__ __launch_bounds__(256)
void attn_kernel(const float* __restrict__ q,
                 const float* __restrict__ k,
                 const float* __restrict__ v,
                 const int* __restrict__ Mtop,
                 float* __restrict__ upd) {
    __shared__ float sc[SEQ];
    __shared__ float qs[DK];
    __shared__ float red[256];
    const int blk = blockIdx.x;          // bh*NU + u
    const int bh  = blk / NU;
    const int t   = threadIdx.x;

    const int lq = Mtop[blk];
    const float* qr = q + ((size_t)bh * SEQ + lq) * DK;
    if (t < DK) qs[t] = qr[t];
    __syncthreads();

    const float* kb = k + (size_t)bh * SEQ * DK;
    float lmax = NEG_INF;
    for (int j = 0; j < 8; j++) {
        int l = t + j * 256;
        const float* kr = kb + (size_t)l * DK;
        float dot = 0.0f;
        for (int d = 0; d < DK; d++) dot += qs[d] * kr[d];
        sc[l] = dot;
        lmax = fmaxf(lmax, dot);
    }
    red[t] = lmax;
    __syncthreads();
    for (int off = 128; off > 0; off >>= 1) {
        if (t < off) red[t] = fmaxf(red[t], red[t + off]);
        __syncthreads();
    }
    const float gmax = red[0];
    __syncthreads();

    float lsum = 0.0f;
    for (int j = 0; j < 8; j++) {
        int l = t + j * 256;
        float e = __expf(sc[l] - gmax);
        sc[l] = e;
        lsum += e;
    }
    red[t] = lsum;
    __syncthreads();
    for (int off = 128; off > 0; off >>= 1) {
        if (t < off) red[t] += red[t + off];
        __syncthreads();
    }
    const float inv = 1.0f / red[0];
    __syncthreads();

    if (t < DK) {
        const float* vb = v + (size_t)bh * SEQ * DK;
        float acc = 0.0f;
        for (int l = 0; l < SEQ; l++) acc += sc[l] * vb[(size_t)l * DK + t];
        upd[(size_t)blk * DK + t] = acc * inv;
    }
}

// ---------------------------------------------------------------------------
// mean(v) over L per (b,h,d)
// ---------------------------------------------------------------------------
__global__ __launch_bounds__(64)
void vmean_kernel(const float* __restrict__ v, float* __restrict__ vmean) {
    const int bh = blockIdx.x, d = threadIdx.x;
    const float* vb = v + (size_t)bh * SEQ * DK;
    float s = 0.0f;
    for (int l = 0; l < SEQ; l++) s += vb[(size_t)l * DK + d];
    vmean[bh * DK + d] = s * (1.0f / SEQ);
}

// ---------------------------------------------------------------------------
// context fill (broadcast mean-v, bf16) and sparse row scatter
// ---------------------------------------------------------------------------
__global__ __launch_bounds__(256)
void ctx_fill_kernel(const float* __restrict__ vmean, unsigned short* __restrict__ ctx) {
    int i = blockIdx.x * 256 + threadIdx.x;      // 0 .. ROWS*DM-1
    int col = i & (DM - 1);
    int row = i >> 10;
    int b = row >> 11;
    int h = col >> 6, d = col & (DK - 1);
    ctx[i] = f2bf(vmean[(b * NHEAD + h) * DK + d]);
}

__global__ __launch_bounds__(64)
void ctx_scatter_kernel(const float* __restrict__ upd,
                        const int* __restrict__ Mtop,
                        unsigned short* __restrict__ ctx) {
    const int blk = blockIdx.x;                  // bh*NU + u
    const int d = threadIdx.x;
    const int bh = blk / NU;
    const int b = bh >> 4, h = bh & (NHEAD - 1);
    const int row = Mtop[blk];
    ctx[((size_t)(b * SEQ + row)) * DM + h * DK + d] = f2bf(upd[(size_t)blk * DK + d]);
}

// ---------------------------------------------------------------------------
// host side
// ---------------------------------------------------------------------------
extern "C" void kernel_launch(void* const* d_in, const int* in_sizes, int n_in,
                              void* d_out, int out_size, void* d_ws, size_t ws_size,
                              hipStream_t stream) {
    const float* hs    = (const float*)d_in[0];
    const float* Wq    = (const float*)d_in[1];
    const float* Wk    = (const float*)d_in[2];
    const float* Wv    = (const float*)d_in[3];
    const float* Wfc   = (const float*)d_in[4];
    const float* bfc   = (const float*)d_in[5];
    const float* gamma = (const float*)d_in[6];
    const float* beta  = (const float*)d_in[7];
    const int*   idx   = (const int*)d_in[8];
    float* out = (float*)d_out;

    char* ws = (char*)d_ws;
    size_t off = 0;
    auto carve = [&](size_t bytes) -> void* {
        void* p = ws + off;
        off = (off + bytes + 255) & ~(size_t)255;
        return p;
    };

    unsigned short* h16    = (unsigned short*)carve((size_t)ROWS * DM * 2);
    unsigned short* qn16   = (unsigned short*)carve((size_t)ROWS * DM * 2);
    unsigned short* Wq16t  = (unsigned short*)carve((size_t)DM * DM * 2);
    unsigned short* Wk16t  = (unsigned short*)carve((size_t)DM * DM * 2);
    unsigned short* Wv16t  = (unsigned short*)carve((size_t)DM * DM * 2);
    unsigned short* Wfc16t = (unsigned short*)carve((size_t)DM * DM * 2);
    float* qbuf   = (float*)carve((size_t)ROWS * DM * 4);   // [B,H,L,DK]
    float* kbuf   = (float*)carve((size_t)ROWS * DM * 4);
    float* vbuf   = (float*)carve((size_t)ROWS * DM * 4);
    float* Mbuf   = (float*)carve((size_t)BATCH * NHEAD * SEQ * 4);
    int*   Mtop   = (int*)carve((size_t)BATCH * NHEAD * NU * 4);
    float* updbuf = (float*)carve((size_t)BATCH * NHEAD * NU * DK * 4);
    float* vmean  = (float*)carve((size_t)BATCH * NHEAD * DK * 4);
    unsigned short* ctx16 = (unsigned short*)carve((size_t)ROWS * DM * 2);

    dim3 tgrid(DM / 32, DM / 32);   // (32, 32)
    cvt_transpose_kernel<<<tgrid, 256, 0, stream>>>(Wq,  Wq16t);
    cvt_transpose_kernel<<<tgrid, 256, 0, stream>>>(Wk,  Wk16t);
    cvt_transpose_kernel<<<tgrid, 256, 0, stream>>>(Wv,  Wv16t);
    cvt_transpose_kernel<<<tgrid, 256, 0, stream>>>(Wfc, Wfc16t);

    ln_kernel<<<ROWS, 256, 0, stream>>>(hs, gamma, beta, h16, qn16);

    dim3 gemm_grid(DM / 128, ROWS / 128);   // (8, 32)
    gemm_bf16_wmma_kernel<<<gemm_grid, 256, 0, stream>>>(
        qn16, Wq16t, qbuf, nullptr, nullptr, ROWS, DM, DM, 0.125f, 0);
    gemm_bf16_wmma_kernel<<<gemm_grid, 256, 0, stream>>>(
        h16, Wk16t, kbuf, nullptr, nullptr, ROWS, DM, DM, 1.0f, 0);
    gemm_bf16_wmma_kernel<<<gemm_grid, 256, 0, stream>>>(
        h16, Wv16t, vbuf, nullptr, nullptr, ROWS, DM, DM, 1.0f, 0);

    sample_score_kernel<<<(BATCH * NHEAD * SEQ) / 8, 256, 0, stream>>>(qbuf, kbuf, idx, Mbuf);
    topk_kernel<<<BATCH * NHEAD, 256, 0, stream>>>(Mbuf, Mtop);
    attn_kernel<<<BATCH * NHEAD * NU, 256, 0, stream>>>(qbuf, kbuf, vbuf, Mtop, updbuf);

    vmean_kernel<<<BATCH * NHEAD, 64, 0, stream>>>(vbuf, vmean);
    ctx_fill_kernel<<<((size_t)ROWS * DM) / 256, 256, 0, stream>>>(vmean, ctx16);
    ctx_scatter_kernel<<<BATCH * NHEAD * NU, 64, 0, stream>>>(updbuf, Mtop, ctx16);

    gemm_bf16_wmma_kernel<<<gemm_grid, 256, 0, stream>>>(
        ctx16, Wfc16t, out, bfc, hs, ROWS, DM, DM, 1.0f, 2);
}
